// ImageLabelModel_79542794322668
// MI455X (gfx1250) — compile-verified, matrix operand order checked by reading, not run on hardware
//
#include <hip/hip_runtime.h>
#include <hip/hip_bf16.h>

// ---------------------------------------------------------------------------
// Transformer decoder (image captioning) for MI455X / gfx1250.
// All GEMMs routed through V_WMMA_F32_16X16X32_F16 (f16 in, f32 accumulate).
// Block tile 128x64, 8 waves, each wave 32x32 (4 WMMAs / K-step).
// K-loop is software-pipelined: global prefetch of tile k+1 overlaps the
// WMMA + LDS-fragment work of tile k.
// ---------------------------------------------------------------------------

#define TILE_N 128
#define TILE_M 64
#define KSTEP  32
#define LDH    40   // padded halves per LDS row (80B, 16B-aligned, scatters banks)

typedef _Float16 h4_t  __attribute__((ext_vector_type(4)));
typedef _Float16 h8_t  __attribute__((ext_vector_type(8)));
typedef _Float16 h16_t __attribute__((ext_vector_type(16)));
typedef float    f8_t  __attribute__((ext_vector_type(8)));

#define FLAG_RELU   1
#define FLAG_CAUSAL 2
#define FLAG_TRANSB 4

struct GemmP {
    const float* A; long lda; long bA0; long bA1;   // A operand (N x K rows)
    const float* B; long ldb; long bB0; long bB1;   // B operand (M x K rows, or K x M if TRANSB)
    const float* bias;                              // per-output-col bias or nullptr
    float*       C; long ldc; long bC0; long bC1;   // output (N x M)
    int N, M, K;
    int Hdim;       // batch z decomposes as (b = z/Hdim, h = z%Hdim)
    float scale;
    int flags;
};

// Build one 16-half WMMA operand fragment from an LDS row (K-contiguous).
// Per ISA 16-bit A/B layout: element e -> k = (e/8)*16 + (lane/16)*8 + (e%8),
// row/col = lane%16.  Two 16-byte LDS reads per lane.
__device__ inline h16_t load_frag(const _Float16* rowp, int g) {
    h8_t lo = *(const h8_t*)(rowp + g * 8);
    h8_t hi = *(const h8_t*)(rowp + 16 + g * 8);
    return __builtin_shufflevector(lo, hi, 0,1,2,3,4,5,6,7,8,9,10,11,12,13,14,15);
}

// ---- phase 1: issue all global loads for one K-step into registers --------
__device__ inline void fetch_tiles(const GemmP& p, const float* A, const float* B,
                                   int row0, int col0, int k0, int tid,
                                   float4* fa, float4* fb) {
    const bool fullK = (k0 + KSTEP) <= p.K;

    if (fullK && (row0 + TILE_N) <= p.N) {
        const float* base = A + (long)row0 * p.lda + k0;
#pragma unroll
        for (int j = 0; j < 4; ++j) {
            int idx = tid + j * 256;              // float4 index, 0..1023
            int r = idx >> 3, c4 = (idx & 7) * 4;
            fa[j] = *(const float4*)(base + (long)r * p.lda + c4);
        }
    } else {
        float* s = (float*)fa;
#pragma unroll
        for (int j = 0; j < 16; ++j) {
            int i = tid + j * 256;
            int r = i >> 5, c = i & 31;
            int gr = row0 + r, gk = k0 + c;
            int rr = gr < p.N ? gr : p.N - 1;
            int kk = gk < p.K ? gk : p.K - 1;
            float v = A[(long)rr * p.lda + kk];
            s[j] = (gr < p.N && gk < p.K) ? v : 0.0f;
        }
    }

    if (!(p.flags & FLAG_TRANSB)) {
        if (fullK && (col0 + TILE_M) <= p.M) {
            const float* base = B + (long)col0 * p.ldb + k0;
#pragma unroll
            for (int j = 0; j < 2; ++j) {
                int idx = tid + j * 256;          // float4 index, 0..511
                int r = idx >> 3, c4 = (idx & 7) * 4;
                fb[j] = *(const float4*)(base + (long)r * p.ldb + c4);
            }
        } else {
            float* s = (float*)fb;
#pragma unroll
            for (int j = 0; j < 8; ++j) {
                int i = tid + j * 256;
                int r = i >> 5, c = i & 31;
                int gm = col0 + r, gk = k0 + c;
                int mm = gm < p.M ? gm : p.M - 1;
                int kk = gk < p.K ? gk : p.K - 1;
                float v = B[(long)mm * p.ldb + kk];
                s[j] = (gm < p.M && gk < p.K) ? v : 0.0f;
            }
        }
    } else {
        if (fullK && (col0 + TILE_M) <= p.M) {
            const float* base = B + (long)k0 * p.ldb + col0;
#pragma unroll
            for (int j = 0; j < 2; ++j) {
                int idx = tid + j * 256;          // float4 index, 0..511
                int kk = idx >> 4, m4 = (idx & 15) * 4;
                fb[j] = *(const float4*)(base + (long)kk * p.ldb + m4);
            }
        } else {
            float* s = (float*)fb;
#pragma unroll
            for (int j = 0; j < 8; ++j) {
                int i = tid + j * 256;
                int kk = i >> 6, mm = i & 63;
                int gk = k0 + kk, gm = col0 + mm;
                int kc = gk < p.K ? gk : p.K - 1;
                int mc = gm < p.M ? gm : p.M - 1;
                float v = B[(long)kc * p.ldb + mc];
                s[j] = (gm < p.M && gk < p.K) ? v : 0.0f;
            }
        }
    }
}

// ---- phase 2: convert f32 -> f16 and store the staged registers to LDS ----
__device__ inline void store_tiles(const GemmP& p, int row0, int col0, int k0, int tid,
                                   const float4* fa, const float4* fb,
                                   _Float16* lA, _Float16* lB) {
    const bool fullK = (k0 + KSTEP) <= p.K;

    if (fullK && (row0 + TILE_N) <= p.N) {
#pragma unroll
        for (int j = 0; j < 4; ++j) {
            int idx = tid + j * 256;
            int r = idx >> 3, c4 = (idx & 7) * 4;
            h4_t h;
            h[0] = (_Float16)fa[j].x; h[1] = (_Float16)fa[j].y;
            h[2] = (_Float16)fa[j].z; h[3] = (_Float16)fa[j].w;
            *(h4_t*)&lA[r * LDH + c4] = h;
        }
    } else {
        const float* s = (const float*)fa;
#pragma unroll
        for (int j = 0; j < 16; ++j) {
            int i = tid + j * 256;
            int r = i >> 5, c = i & 31;
            lA[r * LDH + c] = (_Float16)s[j];
        }
    }

    if (!(p.flags & FLAG_TRANSB)) {
        if (fullK && (col0 + TILE_M) <= p.M) {
#pragma unroll
            for (int j = 0; j < 2; ++j) {
                int idx = tid + j * 256;
                int r = idx >> 3, c4 = (idx & 7) * 4;
                h4_t h;
                h[0] = (_Float16)fb[j].x; h[1] = (_Float16)fb[j].y;
                h[2] = (_Float16)fb[j].z; h[3] = (_Float16)fb[j].w;
                *(h4_t*)&lB[r * LDH + c4] = h;
            }
        } else {
            const float* s = (const float*)fb;
#pragma unroll
            for (int j = 0; j < 8; ++j) {
                int i = tid + j * 256;
                int r = i >> 5, c = i & 31;
                lB[r * LDH + c] = (_Float16)s[j];
            }
        }
    } else {
        if (fullK && (col0 + TILE_M) <= p.M) {
#pragma unroll
            for (int j = 0; j < 2; ++j) {
                int idx = tid + j * 256;
                int kk = idx >> 4, m4 = (idx & 15) * 4;
                lB[(m4 + 0) * LDH + kk] = (_Float16)fb[j].x;
                lB[(m4 + 1) * LDH + kk] = (_Float16)fb[j].y;
                lB[(m4 + 2) * LDH + kk] = (_Float16)fb[j].z;
                lB[(m4 + 3) * LDH + kk] = (_Float16)fb[j].w;
            }
        } else {
            const float* s = (const float*)fb;
#pragma unroll
            for (int j = 0; j < 8; ++j) {
                int i = tid + j * 256;
                int kk = i >> 6, mm = i & 63;
                lB[mm * LDH + kk] = (_Float16)s[j];
            }
        }
    }
}

__global__ __launch_bounds__(256) void wmma_gemm(GemmP p) {
    __shared__ __align__(16) _Float16 lA[TILE_N * LDH];
    __shared__ __align__(16) _Float16 lB[TILE_M * LDH];

    const int z  = blockIdx.z;
    const int bz = z / p.Hdim, hz = z % p.Hdim;
    const float* A = p.A + (long)bz * p.bA0 + (long)hz * p.bA1;
    const float* B = p.B + (long)bz * p.bB0 + (long)hz * p.bB1;
    float*       C = p.C + (long)bz * p.bC0 + (long)hz * p.bC1;

    const int row0 = blockIdx.y * TILE_N;
    const int col0 = blockIdx.x * TILE_M;
    const int tid  = threadIdx.x;
    const int wave = tid >> 5, lane = tid & 31;
    const int wr = (wave >> 1) * 32;   // wave's 32-row strip within tile
    const int wc = (wave & 1) * 32;    // wave's 32-col strip within tile
    const int m16 = lane & 15, g = lane >> 4;

    f8_t acc00 = {}, acc01 = {}, acc10 = {}, acc11 = {};
    float4 fa[4];
    float4 fb[2];

    // prologue: prefetch first K-step
    fetch_tiles(p, A, B, row0, col0, 0, tid, fa, fb);

    for (int k0 = 0; k0 < p.K; k0 += KSTEP) {
        // drain prefetched registers into LDS (f32 -> f16)
        store_tiles(p, row0, col0, k0, tid, fa, fb, lA, lB);
        __syncthreads();

        // issue next K-step's global loads; wait lands at next store_tiles
        if (k0 + KSTEP < p.K)
            fetch_tiles(p, A, B, row0, col0, k0 + KSTEP, tid, fa, fb);

        h16_t af0 = load_frag(&lA[(wr + m16) * LDH], g);
        h16_t af1 = load_frag(&lA[(wr + 16 + m16) * LDH], g);
        h16_t bf0 = load_frag(&lB[(wc + m16) * LDH], g);
        h16_t bf1 = load_frag(&lB[(wc + 16 + m16) * LDH], g);
        acc00 = __builtin_amdgcn_wmma_f32_16x16x32_f16(false, af0, false, bf0,
                                                       (short)0, acc00, false, false);
        acc01 = __builtin_amdgcn_wmma_f32_16x16x32_f16(false, af0, false, bf1,
                                                       (short)0, acc01, false, false);
        acc10 = __builtin_amdgcn_wmma_f32_16x16x32_f16(false, af1, false, bf0,
                                                       (short)0, acc10, false, false);
        acc11 = __builtin_amdgcn_wmma_f32_16x16x32_f16(false, af1, false, bf1,
                                                       (short)0, acc11, false, false);
        __syncthreads();
    }

    // ---- epilogue: scale, bias, relu, causal mask, guarded store ----
    // C layout: v8f element i, lane L -> row = (L/16)*8 + i, col = L%16
#pragma unroll
    for (int ar = 0; ar < 2; ++ar) {
        f8_t a0 = ar ? acc10 : acc00;
        f8_t a1 = ar ? acc11 : acc01;
#pragma unroll
        for (int i = 0; i < 8; ++i) {
            int grow = row0 + wr + ar * 16 + g * 8 + i;
            if (grow >= p.N) continue;
            {
                int gcol = col0 + wc + m16;
                if (gcol < p.M) {
                    float v = a0[i] * p.scale;
                    if (p.bias) v += p.bias[gcol];
                    if (p.flags & FLAG_RELU)   v = fmaxf(v, 0.0f);
                    if ((p.flags & FLAG_CAUSAL) && gcol > grow) v = -3.0e38f;
                    C[(long)grow * p.ldc + gcol] = v;
                }
            }
            {
                int gcol = col0 + wc + 16 + m16;
                if (gcol < p.M) {
                    float v = a1[i] * p.scale;
                    if (p.bias) v += p.bias[gcol];
                    if (p.flags & FLAG_RELU)   v = fmaxf(v, 0.0f);
                    if ((p.flags & FLAG_CAUSAL) && gcol > grow) v = -3.0e38f;
                    C[(long)grow * p.ldc + gcol] = v;
                }
            }
        }
    }
}

// ---------------------------------------------------------------------------
// Row softmax: one 256-thread block per row, length n, row stride ld.
// ---------------------------------------------------------------------------
__global__ __launch_bounds__(256) void softmax_rows(float* S, int n, int ld) {
    long row = blockIdx.x;
    float* r = S + row * (long)ld;
    __shared__ float red[256];
    int tid = threadIdx.x;

    float mx = -3.0e38f;
    for (int i = tid; i < n; i += 256) mx = fmaxf(mx, r[i]);
    red[tid] = mx; __syncthreads();
    for (int s = 128; s > 0; s >>= 1) {
        if (tid < s) red[tid] = fmaxf(red[tid], red[tid + s]);
        __syncthreads();
    }
    mx = red[0]; __syncthreads();

    float sum = 0.0f;
    for (int i = tid; i < n; i += 256) {
        float e = expf(r[i] - mx);
        r[i] = e; sum += e;
    }
    red[tid] = sum; __syncthreads();
    for (int s = 128; s > 0; s >>= 1) {
        if (tid < s) red[tid] += red[tid + s];
        __syncthreads();
    }
    float inv = 1.0f / red[0];
    for (int i = tid; i < n; i += 256) r[i] *= inv;
}

// ---------------------------------------------------------------------------
// Fused residual-add + LayerNorm over D=768: x = LN(x + r)*g + b  (in place)
// ---------------------------------------------------------------------------
__global__ __launch_bounds__(256) void add_ln(float* x, const float* r,
                                              const float* gg, const float* bb) {
    const int Dd = 768;
    long row = blockIdx.x;
    float* xr = x + row * (long)Dd;
    const float* rr = r + row * (long)Dd;
    __shared__ float s1[256], s2[256];

    float v[3], su = 0.0f, sq = 0.0f;
    for (int j = 0; j < 3; ++j) {
        int c = threadIdx.x + j * 256;
        float t = xr[c] + rr[c];
        v[j] = t; su += t; sq += t * t;
    }
    s1[threadIdx.x] = su; s2[threadIdx.x] = sq; __syncthreads();
    for (int s = 128; s > 0; s >>= 1) {
        if (threadIdx.x < s) {
            s1[threadIdx.x] += s1[threadIdx.x + s];
            s2[threadIdx.x] += s2[threadIdx.x + s];
        }
        __syncthreads();
    }
    float mean = s1[0] / Dd;
    float var  = s2[0] / Dd - mean * mean;
    float inv  = rsqrtf(var + 1e-5f);
    for (int j = 0; j < 3; ++j) {
        int c = threadIdx.x + j * 256;
        xr[c] = (v[j] - mean) * inv * gg[c] + bb[c];
    }
}

// ---------------------------------------------------------------------------
// Embedding gather + positional add: x[b,t,:] = tok_emb[labels[b,t]] + pos[t]
// ---------------------------------------------------------------------------
__global__ __launch_bounds__(256) void embed_kernel(const int* labels,
                                                    const float* tok,
                                                    const float* pos,
                                                    float* x) {
    int i = blockIdx.x;          // 0..8191: b = i/512, t = i%512
    int b = i >> 9, t = i & 511;
    int id = labels[b * 513 + t];
    for (int c = threadIdx.x; c < 768; c += 256)
        x[(long)i * 768 + c] = tok[(long)id * 768 + c] + pos[t * 768 + c];
}

// labels[:, 1:] as float into tail of d_out
__global__ __launch_bounds__(256) void labels_tail(const int* labels, float* out) {
    int i = blockIdx.x * 256 + threadIdx.x;
    if (i < 8192) {
        int b = i >> 9, t = i & 511;
        out[i] = (float)labels[b * 513 + t + 1];
    }
}

// ---------------------------------------------------------------------------
// Host side
// ---------------------------------------------------------------------------
static void gemm(hipStream_t st,
                 const float* A, long lda, long bA0, long bA1,
                 const float* B, long ldb, long bB0, long bB1,
                 const float* bias,
                 float* C, long ldc, long bC0, long bC1,
                 int N, int M, int K, int batch, int Hdim,
                 float scale, int flags) {
    GemmP p;
    p.A = A; p.lda = lda; p.bA0 = bA0; p.bA1 = bA1;
    p.B = B; p.ldb = ldb; p.bB0 = bB0; p.bB1 = bB1;
    p.bias = bias;
    p.C = C; p.ldc = ldc; p.bC0 = bC0; p.bC1 = bC1;
    p.N = N; p.M = M; p.K = K; p.Hdim = Hdim; p.scale = scale; p.flags = flags;
    dim3 grid((M + TILE_M - 1) / TILE_M, (N + TILE_N - 1) / TILE_N, batch);
    hipLaunchKernelGGL(wmma_gemm, grid, dim3(256), 0, st, p);
}

extern "C" void kernel_launch(void* const* d_in, const int* in_sizes, int n_in,
                              void* d_out, int out_size, void* d_ws, size_t ws_size,
                              hipStream_t stream) {
    (void)in_sizes; (void)n_in; (void)out_size; (void)ws_size;

    // Model constants
    const int D = 768, H = 4, HD = 192, F = 1024, L = 3, V = 10000;
    const int B = 16, T = 512, S = 197;
    const int N  = B * T;     // 8192 target tokens
    const int NM = B * S;     // 3152 vision tokens
    const float ATT_SCALE = 0.07216878364870322f;  // 1/sqrt(192)

    // Inputs (setup_inputs order)
    const int*   labels     = (const int*)  d_in[0];
    const float* vision     = (const float*)d_in[1];
    const float* vproj_w    = (const float*)d_in[2];
    const float* vproj_b    = (const float*)d_in[3];
    const float* tok_emb    = (const float*)d_in[4];
    const float* pos_emb    = (const float*)d_in[5];
    const float* sa_w       = (const float*)d_in[6];
    const float* sa_b       = (const float*)d_in[7];
    const float* sa_ow      = (const float*)d_in[8];
    const float* sa_ob      = (const float*)d_in[9];
    const float* ca_w       = (const float*)d_in[10];
    const float* ca_b       = (const float*)d_in[11];
    const float* ca_ow      = (const float*)d_in[12];
    const float* ca_ob      = (const float*)d_in[13];
    const float* ln1_g      = (const float*)d_in[14];
    const float* ln1_b      = (const float*)d_in[15];
    const float* ln2_g      = (const float*)d_in[16];
    const float* ln2_b      = (const float*)d_in[17];
    const float* ln3_g      = (const float*)d_in[18];
    const float* ln3_b      = (const float*)d_in[19];
    const float* ff1_w      = (const float*)d_in[20];
    const float* ff1_b      = (const float*)d_in[21];
    const float* ff2_w      = (const float*)d_in[22];
    const float* ff2_b      = (const float*)d_in[23];
    const float* out_w      = (const float*)d_in[24];
    const float* out_b      = (const float*)d_in[25];

    float* OUT = (float*)d_out;

    // Workspace carve-up (floats)
    float* ws = (float*)d_ws;
    float* X      = ws;                         // 8192 x 768
    float* MEMB   = X      + (long)N  * D;      // 3152 x 768
    float* KVC    = MEMB   + (long)NM * D;      // 3152 x 1536
    float* QKV    = KVC    + (long)NM * 2 * D;  // 8192 x 2304
    float* SC     = QKV    + (long)N  * 3 * D;  // 64 x 512 x 512 scores
    float* ATT    = SC     + (long)B * H * T * T;  // 8192 x 768
    float* PRJ    = ATT    + (long)N  * D;      // 8192 x 768
    float* FF1    = PRJ    + (long)N  * D;      // 8192 x 1024

    // 1) embedding + positional, and labels tail of d_out
    hipLaunchKernelGGL(embed_kernel, dim3(N), dim3(256), 0, stream,
                       labels, tok_emb, pos_emb, X);
    hipLaunchKernelGGL(labels_tail, dim3((N + 255) / 256), dim3(256), 0, stream,
                       labels, OUT + (long)N * V);

    // 2) vision memory projection: MEMB = vision @ vproj_w^T + b
    gemm(stream, vision, D, 0, 0, vproj_w, D, 0, 0, vproj_b,
         MEMB, D, 0, 0, NM, D, D, 1, 1, 1.0f, 0);

    for (int l = 0; l < L; ++l) {
        const float* saw  = sa_w  + (long)l * 3 * D * D;
        const float* sab  = sa_b  + (long)l * 3 * D;
        const float* saow = sa_ow + (long)l * D * D;
        const float* saob = sa_ob + (long)l * D;
        const float* caw  = ca_w  + (long)l * 3 * D * D;
        const float* cab  = ca_b  + (long)l * 3 * D;
        const float* caow = ca_ow + (long)l * D * D;
        const float* caob = ca_ob + (long)l * D;

        // ---- self attention ----
        // QKV = X @ saw^T + sab   (8192 x 2304)
        gemm(stream, X, D, 0, 0, saw, D, 0, 0, sab,
             QKV, 3 * D, 0, 0, N, 3 * D, D, 1, 1, 1.0f, 0);
        // scores[b,h] = Q K^T * scale + causal  (512 x 512, K=192)
        gemm(stream, QKV,       3 * D, (long)T * 3 * D, HD,
                     QKV + D,   3 * D, (long)T * 3 * D, HD, nullptr,
             SC, T, (long)H * T * T, (long)T * T,
             T, T, HD, B * H, H, ATT_SCALE, FLAG_CAUSAL);
        hipLaunchKernelGGL(softmax_rows, dim3(B * H * T), dim3(256), 0, stream,
                           SC, T, T);
        // ATT[b,:,h*192:(h+1)*192] = P @ V   (transB: V is K x M rows)
        gemm(stream, SC, T, (long)H * T * T, (long)T * T,
                     QKV + 2 * D, 3 * D, (long)T * 3 * D, HD, nullptr,
             ATT, D, (long)T * D, HD,
             T, HD, T, B * H, H, 1.0f, FLAG_TRANSB);
        // out proj + residual LN
        gemm(stream, ATT, D, 0, 0, saow, D, 0, 0, saob,
             PRJ, D, 0, 0, N, D, D, 1, 1, 1.0f, 0);
        hipLaunchKernelGGL(add_ln, dim3(N), dim3(256), 0, stream,
                           X, PRJ, ln1_g + (long)l * D, ln1_b + (long)l * D);

        // ---- cross attention ----
        // Qc into QKV cols 0..767 (ldc = 3D so score addressing is identical)
        gemm(stream, X, D, 0, 0, caw, D, 0, 0, cab,
             QKV, 3 * D, 0, 0, N, D, D, 1, 1, 1.0f, 0);
        // KV on memory: KVC = MEMB @ caw[D:3D]^T + cab[D:3D]  (3152 x 1536)
        gemm(stream, MEMB, D, 0, 0, caw + (long)D * D, D, 0, 0, cab + D,
             KVC, 2 * D, 0, 0, NM, 2 * D, D, 1, 1, 1.0f, 0);
        // scores (512 x 197, K=192), no mask
        gemm(stream, QKV, 3 * D, (long)T * 3 * D, HD,
                     KVC, 2 * D, (long)S * 2 * D, HD, nullptr,
             SC, T, (long)H * T * T, (long)T * T,
             T, S, HD, B * H, H, ATT_SCALE, 0);
        hipLaunchKernelGGL(softmax_rows, dim3(B * H * T), dim3(256), 0, stream,
                           SC, S, T);
        // ATT = P @ Vc  (transB, K = 197 zero-padded in staging)
        gemm(stream, SC, T, (long)H * T * T, (long)T * T,
                     KVC + D, 2 * D, (long)S * 2 * D, HD, nullptr,
             ATT, D, (long)T * D, HD,
             T, HD, S, B * H, H, 1.0f, FLAG_TRANSB);
        gemm(stream, ATT, D, 0, 0, caow, D, 0, 0, caob,
             PRJ, D, 0, 0, N, D, D, 1, 1, 1.0f, 0);
        hipLaunchKernelGGL(add_ln, dim3(N), dim3(256), 0, stream,
                           X, PRJ, ln2_g + (long)l * D, ln2_b + (long)l * D);

        // ---- feed forward ----
        gemm(stream, X, D, 0, 0, ff1_w + (long)l * F * D, D, 0, 0,
             ff1_b + (long)l * F,
             FF1, F, 0, 0, N, F, D, 1, 1, 1.0f, FLAG_RELU);
        gemm(stream, FF1, F, 0, 0, ff2_w + (long)l * D * F, F, 0, 0,
             ff2_b + (long)l * D,
             PRJ, D, 0, 0, N, D, F, 1, 1, 1.0f, 0);
        hipLaunchKernelGGL(add_ln, dim3(N), dim3(256), 0, stream,
                           X, PRJ, ln3_g + (long)l * D, ln3_b + (long)l * D);
    }

    // 3) logits = X @ out_w^T + out_b  -> d_out (8192 x 10000)
    gemm(stream, X, D, 0, 0, out_w, D, 0, 0, out_b,
         OUT, V, 0, 0, N, V, D, 1, 1, 1.0f, 0);
}
